// RWKV_RNN_70411693850780
// MI455X (gfx1250) — matrix operand order repaired and checked by reading, not compiled, not run on hardware
//
#include <hip/hip_runtime.h>
#include <hip/hip_bf16.h>
#include <cstdint>

// ---------------------------------------------------------------------------
// RWKV forward for MI455X (gfx1250):
//   - weights/activations pre-converted to bf16 (K-contiguous for both A & B)
//   - GEMM stages tiles with global_load_async_to_lds_b128 (ASYNCcnt),
//     double-buffered LDS, fragments via ds_load_b128, v_wmma bf16 math
//   - WKV scan one lane per channel
// ---------------------------------------------------------------------------

typedef __attribute__((ext_vector_type(16))) __bf16 v16bf;
typedef __attribute__((ext_vector_type(8)))  float  v8f;

static constexpr int Lr  = 12;
static constexpr int Dm  = 1024;
static constexpr int Vv  = 50277;
static constexpr int Tt  = 1024;
static constexpr int F4m = 4096;

__device__ __forceinline__ unsigned short bf16_bits(float x) {
  union { __bf16 b; unsigned short u; } q;
  q.b = (__bf16)x;
  return q.u;
}

// async copy of 16B per lane, global -> LDS (GLOBAL_LOAD_ASYNC_TO_LDS_B128)
#define ASYNC_CP16(lds_u32, gsrc)                                   \
  asm volatile("global_load_async_to_lds_b128 %0, %1, off"          \
               :: "v"(lds_u32), "v"(gsrc) : "memory")

// ---------------------------------------------------------------------------
// WMMA GEMM: C[M,N] = epilogue(A[M,K] @ B^T[N,K])   (both operands bf16)
//
//  Block: 256 threads = 8 waves, tile 64(M) x 128(N), K-step 32.
//  Waves 4(M) x 2(N); each wave computes 16x64 (4 accumulators, A reused).
//  Per K-tile each wave issues 3 async b128 loads (A:1, B:2); tiles are
//  double-buffered: tile k+1 streams into LDS while tile k feeds the WMMAs
//  (wait asynccnt<=3 keeps the newest 3 in flight).
//  All 10 fragment ds_load_b128 are issued before the 4 WMMAs so the later
//  loads drain underneath the earlier WMMAs (partial dscnt waits).
//  Fragment layouts per ISA 7.12.2:
//    A frag lane l: row M=l&15, halves K = kbA+{0..7},{16..23}, kbA=8*(l>=16)
//    B frag lane l: col N=l&15, halves K = kbB+{0..7},{16..23}, kbB=8*(l>=16)
//
//  mode: 0 plain | 1 sigmoid | 2 relu^2 (bf16 out to Cbf) | 3 residual+P
//        | 4 residual+gate*P
// ---------------------------------------------------------------------------
__global__ __launch_bounds__(256) void gemm_bf16_wmma(
    const unsigned short* __restrict__ A,    // [M,K] bf16
    const unsigned short* __restrict__ B,    // [N,K] bf16 (pre-transposed)
    float* __restrict__ C, unsigned short* __restrict__ Cbf,
    int M_, int K_, int N_,
    const float* __restrict__ residual, const float* __restrict__ gate,
    int mode)
{
  // row pitch 40 halves = 80B: 16B-aligned rows, stride-20-dword banking
  __shared__ __align__(16) unsigned short lA[2][64][40];
  __shared__ __align__(16) unsigned short lB[2][128][40];

  const int tid    = threadIdx.x;
  const int lane   = tid & 31;
  const int wid    = tid >> 5;
  const int lhalf  = lane >> 4;
  const int lidx   = lane & 15;
  const int wave_m = wid >> 1;   // 0..3
  const int wave_n = wid & 1;    // 0..1
  const int n0 = blockIdx.x * 128;
  const int m0 = blockIdx.y * 64;

  const int kbA = lhalf * 8;
  const int kbB = lhalf * 8;

  const v8f vz = {0.f, 0.f, 0.f, 0.f, 0.f, 0.f, 0.f, 0.f};
  v8f acc[4] = {vz, vz, vz, vz};

  // 16B chunk assignment: A tile 64 rows x 4 chunks, B tile 128 rows x 4
  const int achunk  = wid * 32 + lane;        // 256 chunks
  const int bchunk0 = wid * 64 + lane;        // 512 chunks
  const int bchunk1 = wid * 64 + 32 + lane;

  const int ar  = achunk  >> 2, ac  = (achunk  & 3) * 8;
  const int br0 = bchunk0 >> 2, bc0 = (bchunk0 & 3) * 8;
  const int br1 = bchunk1 >> 2, bc1 = (bchunk1 & 3) * 8;

  auto issue_tile = [&](int buf, int k0) {
    {
      const unsigned short* src = A + (size_t)(m0 + ar) * (size_t)K_ + k0 + ac;
      const unsigned dst = (unsigned)(uintptr_t)&lA[buf][ar][ac];
      ASYNC_CP16(dst, src);
    }
    {
      const unsigned short* src = B + (size_t)(n0 + br0) * (size_t)K_ + k0 + bc0;
      const unsigned dst = (unsigned)(uintptr_t)&lB[buf][br0][bc0];
      ASYNC_CP16(dst, src);
    }
    {
      const unsigned short* src = B + (size_t)(n0 + br1) * (size_t)K_ + k0 + bc1;
      const unsigned dst = (unsigned)(uintptr_t)&lB[buf][br1][bc1];
      ASYNC_CP16(dst, src);
    }
  };

  union FragU { uint4 q[2]; v16bf v; };

  const int ntiles = K_ >> 5;
  issue_tile(0, 0);
  for (int it = 0; it < ntiles; ++it) {
    const int cur = it & 1;
    if (it + 1 < ntiles) {
      issue_tile(cur ^ 1, (it + 1) * 32);      // stream next tile
      asm volatile("s_wait_asynccnt 0x3" ::: "memory");  // tile `it` landed
    } else {
      asm volatile("s_wait_asynccnt 0x0" ::: "memory");
    }
    __syncthreads();                           // tile visible to all waves

    // issue ALL fragment loads first (in-order DS returns -> partial waits)
    FragU fa;
    const unsigned short* ap = &lA[cur][wave_m * 16 + lidx][0];
    fa.q[0] = *(const uint4*)(ap + kbA);
    fa.q[1] = *(const uint4*)(ap + kbA + 16);

    FragU fb[4];
#pragma unroll
    for (int nt = 0; nt < 4; ++nt) {
      const unsigned short* bp = &lB[cur][wave_n * 64 + nt * 16 + lidx][0];
      fb[nt].q[0] = *(const uint4*)(bp + kbB);
      fb[nt].q[1] = *(const uint4*)(bp + kbB + 16);
    }

#pragma unroll
    for (int nt = 0; nt < 4; ++nt) {
      acc[nt] = __builtin_amdgcn_wmma_f32_16x16x32_bf16(
          false, fa.v, false, fb[nt].v, (short)0, acc[nt], false, false);
    }
    __syncthreads();                           // done reading buffer `cur`
  }

  // C/D layout: VGPR r -> M = r + 8*lhalf, N = lane&15
#pragma unroll
  for (int nt = 0; nt < 4; ++nt) {
    const int nn = n0 + wave_n * 64 + nt * 16 + lidx;
#pragma unroll
    for (int r = 0; r < 8; ++r) {
      const int mm = m0 + wave_m * 16 + r + lhalf * 8;
      const size_t idx = (size_t)mm * (size_t)N_ + nn;
      float v = acc[nt][r];
      if (mode == 2) {                         // relu^2, bf16 out (feeds GEMM)
        v = fmaxf(v, 0.f);
        Cbf[idx] = bf16_bits(v * v);
      } else {
        if (mode == 1)      { v = 1.f / (1.f + __expf(-v)); }
        else if (mode == 3) { v = residual[idx] + v; }
        else if (mode == 4) { v = residual[idx] + gate[idx] * v; }
        C[idx] = v;
      }
    }
  }
}

// ---------------------------------------------------------------------------
// Weight convert + transpose: in[K,N] f32 -> out[N,K] bf16 (32x32 LDS tiles)
// ---------------------------------------------------------------------------
__global__ __launch_bounds__(256) void transp_bf16(
    const float* __restrict__ in, unsigned short* __restrict__ out,
    int K_, int N_)
{
  __shared__ float t[32][33];
  const int tx = threadIdx.x;        // 0..31
  const int ty = threadIdx.y;        // 0..7
  const int nb = blockIdx.x * 32;
  const int kb = blockIdx.y * 32;
#pragma unroll
  for (int i = 0; i < 4; ++i) {
    const int k = kb + ty + i * 8;
    t[ty + i * 8][tx] = in[(size_t)k * (size_t)N_ + nb + tx];
  }
  __syncthreads();
#pragma unroll
  for (int i = 0; i < 4; ++i) {
    const int n = nb + ty + i * 8;
    out[(size_t)n * (size_t)K_ + kb + tx] = bf16_bits(t[tx][ty + i * 8]);
  }
}

// ---------------------------------------------------------------------------
// LayerNorm over rows: one block (256 thr = 8 waves) per row.
// ---------------------------------------------------------------------------
__device__ __forceinline__ float wave_sum(float v) {
#pragma unroll
  for (int off = 16; off > 0; off >>= 1) v += __shfl_xor(v, off, 32);
  return v;
}

__device__ __forceinline__ void ln_row_body(
    const float* __restrict__ xr, const float* __restrict__ w,
    const float* __restrict__ b, float* __restrict__ yr, int Dd,
    float* sA, float* sB)
{
  float s = 0.f, s2 = 0.f;
  for (int d = threadIdx.x; d < Dd; d += blockDim.x) {
    const float v = xr[d];
    s += v; s2 += v * v;
  }
  const int wid = threadIdx.x >> 5, lane = threadIdx.x & 31;
  s  = wave_sum(s);
  s2 = wave_sum(s2);
  if (lane == 0) { sA[wid] = s; sB[wid] = s2; }
  __syncthreads();
  if (threadIdx.x == 0) {
    float a = 0.f, c = 0.f;
#pragma unroll
    for (int i = 0; i < 8; ++i) { a += sA[i]; c += sB[i]; }
    sA[0] = a; sB[0] = c;
  }
  __syncthreads();
  const float mu  = sA[0] / (float)Dd;
  const float var = sB[0] / (float)Dd - mu * mu;
  const float inv = rsqrtf(var + 1e-5f);
  for (int d = threadIdx.x; d < Dd; d += blockDim.x) {
    yr[d] = (xr[d] - mu) * inv * w[d] + b[d];
  }
}

__global__ __launch_bounds__(256) void ln_rows(
    const float* __restrict__ x, const float* __restrict__ w,
    const float* __restrict__ b, float* __restrict__ y, int Dd)
{
  __shared__ float sA[8];
  __shared__ float sB[8];
  const int t = blockIdx.x;
  ln_row_body(x + (size_t)t * Dd, w, b, y + (size_t)t * Dd, Dd, sA, sB);
}

__global__ __launch_bounds__(256) void embed_ln0(
    const int* __restrict__ tokens, const float* __restrict__ emb,
    const float* __restrict__ w, const float* __restrict__ b,
    float* __restrict__ y, int Dd)
{
  __shared__ float sA[8];
  __shared__ float sB[8];
  const int t = blockIdx.x;
  ln_row_body(emb + (size_t)tokens[t] * Dd, w, b, y + (size_t)t * Dd, Dd, sA, sB);
}

// ---------------------------------------------------------------------------
// Token shift + mixing (emit bf16: these feed GEMM A operands only)
// ---------------------------------------------------------------------------
__global__ __launch_bounds__(256) void shiftmix3(
    const float* __restrict__ xn,
    const float* __restrict__ tmk, const float* __restrict__ tmv,
    const float* __restrict__ tmr,
    unsigned short* __restrict__ xk, unsigned short* __restrict__ xv,
    unsigned short* __restrict__ xr, int Ttot, int Dd)
{
  const int i = blockIdx.x * blockDim.x + threadIdx.x;
  if (i >= Ttot * Dd) return;
  const int t = i / Dd, d = i - t * Dd;
  const float cur  = xn[i];
  const float prev = (t > 0) ? xn[i - Dd] : 0.f;
  const float mk = tmk[d], mv = tmv[d], mr = tmr[d];
  xk[i] = bf16_bits(cur * mk + prev * (1.f - mk));
  xv[i] = bf16_bits(cur * mv + prev * (1.f - mv));
  xr[i] = bf16_bits(cur * mr + prev * (1.f - mr));
}

__global__ __launch_bounds__(256) void shiftmix2(
    const float* __restrict__ xn,
    const float* __restrict__ tmk, const float* __restrict__ tmr,
    unsigned short* __restrict__ xk, unsigned short* __restrict__ xr,
    int Ttot, int Dd)
{
  const int i = blockIdx.x * blockDim.x + threadIdx.x;
  if (i >= Ttot * Dd) return;
  const int t = i / Dd, d = i - t * Dd;
  const float cur  = xn[i];
  const float prev = (t > 0) ? xn[i - Dd] : 0.f;
  const float mk = tmk[d], mr = tmr[d];
  xk[i] = bf16_bits(cur * mk + prev * (1.f - mk));
  xr[i] = bf16_bits(cur * mr + prev * (1.f - mr));
}

// ---------------------------------------------------------------------------
// WKV recurrence: one lane per channel, sequential over T (coalesced k/v).
// ---------------------------------------------------------------------------
__global__ __launch_bounds__(256) void wkv_scan(
    const float* __restrict__ k, const float* __restrict__ v,
    const float* __restrict__ tf, const float* __restrict__ td,
    float* __restrict__ wkv, int Ttot, int Dd)
{
  const int d = blockIdx.x * blockDim.x + threadIdx.x;
  if (d >= Dd) return;
  const float tfd = tf[d], tdd = td[d];
  float aa = 0.f, bb = 0.f, pp = -1e30f;
  for (int t = 0; t < Ttot; ++t) {
    const size_t i = (size_t)t * Dd + d;
    const float kt = k[i], vt = v[i];
    float ww = tfd + kt;
    float p  = fmaxf(pp, ww);
    float e1 = expf(pp - p);
    float e2 = expf(ww - p);
    wkv[i] = (e1 * aa + e2 * vt) / (e1 * bb + e2);
    ww = pp + tdd;
    p  = fmaxf(ww, kt);
    e1 = expf(ww - p);
    e2 = expf(kt - p);
    aa = e1 * aa + e2 * vt;
    bb = e1 * bb + e2;
    pp = p;
  }
}

// rwkv = sigmoid(r) * wkv, emitted bf16 (feeds output-projection GEMM)
__global__ __launch_bounds__(256) void elem_mul_bf16(
    const float* __restrict__ a, const float* __restrict__ b,
    unsigned short* __restrict__ c, int n)
{
  const int i = blockIdx.x * blockDim.x + threadIdx.x;
  if (i < n) c[i] = bf16_bits(a[i] * b[i]);
}

// ---------------------------------------------------------------------------
// Head GEMV: logits[V] = head[V,D] @ xo[D]  (one wave per row, float4 loads)
// ---------------------------------------------------------------------------
__global__ __launch_bounds__(256) void head_gemv(
    const float* __restrict__ head, const float* __restrict__ xo,
    float* __restrict__ out, int V_, int D_)
{
  const int lane = threadIdx.x & 31;
  const int row  = blockIdx.x * (blockDim.x >> 5) + (threadIdx.x >> 5);
  if (row >= V_) return;
  const float* hr = head + (size_t)row * D_;
  float sum = 0.f;
  for (int d = lane * 4; d < D_; d += 128) {
    const float4 h = *(const float4*)(hr + d);
    const float4 x = *(const float4*)(xo + d);
    sum += h.x * x.x + h.y * x.y + h.z * x.z + h.w * x.w;
  }
  sum = wave_sum(sum);
  if (lane == 0) out[row] = sum;
}

// ---------------------------------------------------------------------------
// Orchestration
// ---------------------------------------------------------------------------
extern "C" void kernel_launch(void* const* d_in, const int* in_sizes, int n_in,
                              void* d_out, int out_size, void* d_ws, size_t ws_size,
                              hipStream_t stream) {
  const int*   tokens     = (const int*)  d_in[0];
  const float* emb        = (const float*)d_in[1];
  const float* ln0_w      = (const float*)d_in[2];
  const float* ln0_b      = (const float*)d_in[3];
  const float* ln1_w      = (const float*)d_in[4];
  const float* ln1_b      = (const float*)d_in[5];
  const float* ln2_w      = (const float*)d_in[6];
  const float* ln2_b      = (const float*)d_in[7];
  const float* tmk        = (const float*)d_in[8];
  const float* tmv        = (const float*)d_in[9];
  const float* tmr        = (const float*)d_in[10];
  const float* tmk_ffn    = (const float*)d_in[11];
  const float* tmr_ffn    = (const float*)d_in[12];
  const float* time_first = (const float*)d_in[13];
  const float* time_decay = (const float*)d_in[14];
  const float* att_kw     = (const float*)d_in[15];
  const float* att_vw     = (const float*)d_in[16];
  const float* att_rw     = (const float*)d_in[17];
  const float* att_ow     = (const float*)d_in[18];
  const float* ffn_kw     = (const float*)d_in[19];
  const float* ffn_vw     = (const float*)d_in[20];
  const float* ffn_rw     = (const float*)d_in[21];
  const float* lnout_w    = (const float*)d_in[22];
  const float* lnout_b    = (const float*)d_in[23];
  const float* head       = (const float*)d_in[24];
  float* logits = (float*)d_out;

  // Workspace carve (256B aligned slices)
  const size_t TD = (size_t)Tt * Dm;
  char* p = (char*)d_ws;
  auto alloc = [&](size_t bytes) -> char* {
    char* r = p;
    p += (bytes + 255) & ~(size_t)255;
    return r;
  };
  float*          x      = (float*)         alloc(TD * 4);
  float*          xn     = (float*)         alloc(TD * 4);
  float*          kbuf   = (float*)         alloc(TD * 4);
  float*          vbuf   = (float*)         alloc(TD * 4);
  float*          rbuf   = (float*)         alloc(TD * 4);
  float*          wkvb   = (float*)         alloc(TD * 4);
  unsigned short* xk_b   = (unsigned short*)alloc(TD * 2);
  unsigned short* xv_b   = (unsigned short*)alloc(TD * 2);
  unsigned short* xr_b   = (unsigned short*)alloc(TD * 2);
  unsigned short* rwkv_b = (unsigned short*)alloc(TD * 2);
  unsigned short* kf_b   = (unsigned short*)alloc(TD * 2 * 4);
  unsigned short* wT     = (unsigned short*)alloc((size_t)Dm * F4m * 2);
  float*          xo     = (float*)         alloc((size_t)Dm * 4);

  const dim3 blk256(256);
  const dim3 blkT(32, 8);
  const dim3 grid_e((unsigned)((TD + 255) / 256));
  const dim3 gemm_DD(Dm / 128, Tt / 64);
  const dim3 gemm_DF(F4m / 128, Tt / 64);
  const dim3 gemm_FD(Dm / 128, Tt / 64);
  const dim3 tr_DD(Dm / 32, Dm / 32);
  const dim3 tr_DF(F4m / 32, Dm / 32);    // in [D,F4] -> out [F4,D]
  const dim3 tr_FD(Dm / 32, F4m / 32);    // in [F4,D] -> out [D,F4]

  // x = ln0(emb[tokens])
  embed_ln0<<<Tt, blk256, 0, stream>>>(tokens, emb, ln0_w, ln0_b, x, Dm);

  for (int l = 0; l < Lr; ++l) {
    const size_t lD  = (size_t)l * Dm;
    const size_t lDD = (size_t)l * Dm * Dm;
    const size_t lDF = (size_t)l * Dm * F4m;

    // ---- time-mix (attention) ----
    ln_rows<<<Tt, blk256, 0, stream>>>(x, ln1_w + lD, ln1_b + lD, xn, Dm);
    shiftmix3<<<grid_e, blk256, 0, stream>>>(xn, tmk + lD, tmv + lD, tmr + lD,
                                             xk_b, xv_b, xr_b, Tt, Dm);
    transp_bf16<<<tr_DD, blkT, 0, stream>>>(att_rw + lDD, wT, Dm, Dm);
    gemm_bf16_wmma<<<gemm_DD, blk256, 0, stream>>>(
        xr_b, wT, rbuf, nullptr, Tt, Dm, Dm, nullptr, nullptr, 1); // sigmoid
    transp_bf16<<<tr_DD, blkT, 0, stream>>>(att_kw + lDD, wT, Dm, Dm);
    gemm_bf16_wmma<<<gemm_DD, blk256, 0, stream>>>(
        xk_b, wT, kbuf, nullptr, Tt, Dm, Dm, nullptr, nullptr, 0);
    transp_bf16<<<tr_DD, blkT, 0, stream>>>(att_vw + lDD, wT, Dm, Dm);
    gemm_bf16_wmma<<<gemm_DD, blk256, 0, stream>>>(
        xv_b, wT, vbuf, nullptr, Tt, Dm, Dm, nullptr, nullptr, 0);
    wkv_scan<<<Dm / 256, blk256, 0, stream>>>(kbuf, vbuf, time_first + lD,
                                              time_decay + lD, wkvb, Tt, Dm);
    elem_mul_bf16<<<grid_e, blk256, 0, stream>>>(rbuf, wkvb, rwkv_b, (int)TD);
    transp_bf16<<<tr_DD, blkT, 0, stream>>>(att_ow + lDD, wT, Dm, Dm);
    gemm_bf16_wmma<<<gemm_DD, blk256, 0, stream>>>(
        rwkv_b, wT, x, nullptr, Tt, Dm, Dm, x, nullptr, 3);       // x += .@ow

    // ---- channel-mix (ffn) ----
    ln_rows<<<Tt, blk256, 0, stream>>>(x, ln2_w + lD, ln2_b + lD, xn, Dm);
    shiftmix2<<<grid_e, blk256, 0, stream>>>(xn, tmk_ffn + lD, tmr_ffn + lD,
                                             xk_b, xr_b, Tt, Dm);
    transp_bf16<<<tr_DD, blkT, 0, stream>>>(ffn_rw + lDD, wT, Dm, Dm);
    gemm_bf16_wmma<<<gemm_DD, blk256, 0, stream>>>(
        xr_b, wT, rbuf, nullptr, Tt, Dm, Dm, nullptr, nullptr, 1); // sigmoid
    transp_bf16<<<tr_DF, blkT, 0, stream>>>(ffn_kw + lDF, wT, Dm, F4m);
    gemm_bf16_wmma<<<gemm_DF, blk256, 0, stream>>>(
        xk_b, wT, nullptr, kf_b, Tt, Dm, F4m, nullptr, nullptr, 2); // relu^2
    transp_bf16<<<tr_FD, blkT, 0, stream>>>(ffn_vw + lDF, wT, F4m, Dm);
    gemm_bf16_wmma<<<gemm_FD, blk256, 0, stream>>>(
        kf_b, wT, x, nullptr, Tt, F4m, Dm, x, rbuf, 4);           // x += r*(.@vw)
  }

  // final LN on last token, then logits = head @ xo
  ln_rows<<<1, blk256, 0, stream>>>(x + (size_t)(Tt - 1) * Dm, lnout_w, lnout_b,
                                    xo, Dm);
  head_gemv<<<(Vv + 7) / 8, blk256, 0, stream>>>(head, xo, logits, Vv, Dm);
}
